// Attention_70772471103641
// MI455X (gfx1250) — matrix operand order repaired and checked by reading, not compiled, not run on hardware
//
#include <hip/hip_runtime.h>
#include <hip/hip_bf16.h>

typedef __attribute__((ext_vector_type(16))) __bf16 v16bf;
typedef __attribute__((ext_vector_type(8)))  float  v8f;

constexpr int kB  = 2;
constexpr int kC  = 256;   // channels
constexpr int kTD = 768;   // heads*features
constexpr int kO3 = 2304;  // 3*kTD
constexpr int kHW = 256;   // spatial positions
#define EPSV 1e-5f

union FragBF { v16bf v; unsigned int u[8]; };

__device__ __forceinline__ unsigned short f2bf(float f) {
  unsigned int u = __float_as_uint(f);
  u += 0x7FFFu + ((u >> 16) & 1u);   // round-to-nearest-even
  return (unsigned short)(u >> 16);
}

// Two 128-bit loads -> one 16x32 bf16 fragment (A or B, both are two
// contiguous 8-ushort runs per lane once operands are stored K-major).
__device__ __forceinline__ FragBF load2x128(const unsigned short* p0,
                                            const unsigned short* p1) {
  FragBF f;
  uint4 a = *(const uint4*)p0;
  uint4 b = *(const uint4*)p1;
  f.u[0] = a.x; f.u[1] = a.y; f.u[2] = a.z; f.u[3] = a.w;
  f.u[4] = b.x; f.u[5] = b.y; f.u[6] = b.z; f.u[7] = b.w;
  return f;
}

__device__ __forceinline__ v8f wmma_bf16(const FragBF& A, const FragBF& B, v8f c) {
  return __builtin_amdgcn_wmma_f32_16x16x32_bf16(false, A.v, false, B.v,
                                                 (short)0, c, false, false);
}

// ---------------------------------------------------------------- converts
__global__ void convert_bf16(const float* __restrict__ in,
                             unsigned short* __restrict__ out, int n) {
  int i = blockIdx.x * blockDim.x + threadIdx.x;
  if (i < n) out[i] = f2bf(in[i]);
}

// x (b,c,p) -> bf16 transposed (b,p,c) so the QKV GEMM's B operand is K-major
__global__ void convert_x_T(const float* __restrict__ in,
                            unsigned short* __restrict__ out) {
  int i = blockIdx.x * blockDim.x + threadIdx.x;   // kB*kC*kHW
  int b = i >> 16;
  int c = (i >> 8) & 255;
  int p = i & 255;
  out[((size_t)b * kHW + p) * kC + c] = f2bf(in[i]);
}

// ---------------------------------------------------------------- QKV GEMM
// QKV[b][o][p] = sum_c Wqkv[o][c] * XT[b][p][c]
// One wave per 16x32 (o,p) tile pair: shared A fragment, two WMMAs per K step.
__global__ void qkv_gemm(const unsigned short* __restrict__ Wb,   // [kO3][kC] bf16
                         const unsigned short* __restrict__ XT,   // [kB][kHW][kC] bf16
                         float* __restrict__ QKV) {               // [kB][kO3][kHW] f32
  const int wave = threadIdx.x >> 5;
  const int lane = threadIdx.x & 31;
  const int col  = lane & 15;
  const int half = lane >> 4;

  int wid = blockIdx.x * 8 + wave;                  // 2304 waves total
  const int mTiles = kO3 / 16;                      // 144
  int b   = wid / (mTiles * 8);
  int rem = wid % (mTiles * 8);
  int o0  = (rem / 8) * 16;
  int p0  = (rem % 8) * 32;

  const unsigned short* Arow = Wb + (size_t)(o0 + col) * kC;
  const unsigned short* B0   = XT + ((size_t)b * kHW + p0 + col) * kC;
  const unsigned short* B1   = XT + ((size_t)b * kHW + p0 + 16 + col) * kC;
  const int aOff = 8 * half;       // A: K run  {2*aOff..+15}
  const int bOff = 16 * half;      // B: K run  {bOff..+15}

  v8f acc0 = {}, acc1 = {};
#pragma unroll 2
  for (int c0 = 0; c0 < kC; c0 += 32) {
    FragBF A  = load2x128(Arow + c0 + 2 * aOff, Arow + c0 + 2 * aOff + 8);
    FragBF Bf0 = load2x128(B0 + c0 + bOff, B0 + c0 + bOff + 8);
    FragBF Bf1 = load2x128(B1 + c0 + bOff, B1 + c0 + bOff + 8);
    acc0 = wmma_bf16(A, Bf0, acc0);
    acc1 = wmma_bf16(A, Bf1, acc1);
  }
  float* out = QKV + ((size_t)b * kO3 + o0) * kHW + p0;
#pragma unroll
  for (int rr = 0; rr < 8; ++rr) {
    out[(rr + 8 * half) * kHW + col]      = acc0[rr];
    out[(rr + 8 * half) * kHW + 16 + col] = acc1[rr];
  }
}

// ---------------------------------------------------------------- attention
// Per (b, head*feat): out_i = sum_j softmax_j(q_i*k_j*s) * v_j   (rank-1 logits)
// Output written TRANSPOSED [b][p][o] bf16 so the projection GEMM B is K-major.
__global__ void attn_softmax(const float* __restrict__ QKV,
                             unsigned short* __restrict__ AOT) {  // [kB][kHW][kTD]
  int g  = blockIdx.x;            // 0 .. kB*kTD-1
  int b  = g / kTD;
  int nd = g % kTD;
  const float* qp = QKV + ((size_t)b * kO3 + nd) * kHW;
  const float* kp = QKV + ((size_t)b * kO3 + kTD + nd) * kHW;
  const float* vp = QKV + ((size_t)b * kO3 + 2 * kTD + nd) * kHW;

  __shared__ float ks[kHW], vs[kHW], rmx[kHW], rmn[kHW];
  int t = threadIdx.x;
  float kv = kp[t];
  ks[t] = kv; vs[t] = vp[t];
  rmx[t] = kv; rmn[t] = kv;
  __syncthreads();
  for (int s = 128; s > 0; s >>= 1) {
    if (t < s) {
      rmx[t] = fmaxf(rmx[t], rmx[t + s]);
      rmn[t] = fminf(rmn[t], rmn[t + s]);
    }
    __syncthreads();
  }
  const float scale = 0.125f;                 // 64^-0.5
  float a = qp[t] * scale;
  float m = (a >= 0.f) ? a * rmx[0] : a * rmn[0];   // row max of rank-1 logits
  float S = 0.f, N = 0.f;
#pragma unroll 4
  for (int j = 0; j < kHW; ++j) {
    float e = __expf(a * ks[j] - m);
    S += e;
    N += e * vs[j];
  }
  AOT[((size_t)b * kHW + t) * kTD + nd] = f2bf(N / S);
}

// ---------------------------------------------------------------- proj + residual + instance norm
// y[b][c][p] = x + b_out[c] + sum_o Wout[c][o]*AOT[b][p][o], then per-(b,c) LN over p
__global__ void proj_norm(const unsigned short* __restrict__ Wob,  // [kC][kTD] bf16
                          const unsigned short* __restrict__ AOT,  // [kB][kHW][kTD] bf16
                          const float* __restrict__ X,
                          const float* __restrict__ bout,
                          float* __restrict__ Y) {
  const int wave = threadIdx.x >> 5;
  const int lane = threadIdx.x & 31;
  const int col  = lane & 15;
  const int half = lane >> 4;
  int b  = blockIdx.x >> 4;
  int c0 = (blockIdx.x & 15) * 16;

  const unsigned short* Arow = Wob + (size_t)(c0 + col) * kTD;
  const int p0   = wave * 32;                       // 8 waves cover all 256 p
  const unsigned short* B0 = AOT + ((size_t)b * kHW + p0 + col) * kTD;
  const unsigned short* B1 = AOT + ((size_t)b * kHW + p0 + 16 + col) * kTD;
  const int aOff = 8 * half;
  const int bOff = 16 * half;

  __shared__ float yS[16][kHW];
  __shared__ float ps[16][16], pq[16][16];
  __shared__ float mu[16], rs[16];

  v8f acc0 = {}, acc1 = {};
#pragma unroll 2
  for (int o0 = 0; o0 < kTD; o0 += 32) {
    FragBF A   = load2x128(Arow + o0 + 2 * aOff, Arow + o0 + 2 * aOff + 8);
    FragBF Bf0 = load2x128(B0 + o0 + bOff, B0 + o0 + bOff + 8);
    FragBF Bf1 = load2x128(B1 + o0 + bOff, B1 + o0 + bOff + 8);
    acc0 = wmma_bf16(A, Bf0, acc0);
    acc1 = wmma_bf16(A, Bf1, acc1);
  }
#pragma unroll
  for (int rr = 0; rr < 8; ++rr) {
    int row = rr + 8 * half;
    int c   = c0 + row;
    float bo = bout[c];
    const float* xr = X + ((size_t)b * kC + c) * kHW;
    yS[row][p0 + col]      = acc0[rr] + xr[p0 + col] + bo;
    yS[row][p0 + 16 + col] = acc1[rr] + xr[p0 + 16 + col] + bo;
  }
  __syncthreads();

  // instance-norm over the 256 spatial positions of each of the 16 rows
  int t = threadIdx.x;
  int r = t >> 4, l = t & 15;
  float s = 0.f, q = 0.f;
#pragma unroll
  for (int k = 0; k < 16; ++k) {
    float v = yS[r][l + (k << 4)];
    s += v; q += v * v;
  }
  ps[r][l] = s; pq[r][l] = q;
  __syncthreads();
  if (t < 16) {
    float st = 0.f, qt = 0.f;
#pragma unroll
    for (int k = 0; k < 16; ++k) { st += ps[t][k]; qt += pq[t][k]; }
    float m   = st * (1.f / 256.f);
    float var = qt * (1.f / 256.f) - m * m;
    mu[t] = m;
    rs[t] = rsqrtf(var + EPSV);
  }
  __syncthreads();
  for (int idx = t; idx < 16 * kHW; idx += 256) {
    int rr = idx >> 8, pp = idx & 255;
    Y[((size_t)b * kC + c0 + rr) * kHW + pp] = (yS[rr][pp] - mu[rr]) * rs[rr];
  }
}

// ---------------------------------------------------------------- launch
extern "C" void kernel_launch(void* const* d_in, const int* in_sizes, int n_in,
                              void* d_out, int out_size, void* d_ws, size_t ws_size,
                              hipStream_t stream) {
  const float* x    = (const float*)d_in[0];  // (2,256,16,16)
  const float* wqkv = (const float*)d_in[1];  // (2304,256)
  const float* wout = (const float*)d_in[2];  // (256,768)
  const float* bout = (const float*)d_in[3];  // (256,)
  float* y = (float*)d_out;

  char* ws = (char*)d_ws;
  unsigned short* Wb  = (unsigned short*)(ws);                     // 2304*256 bf16
  unsigned short* XT  = (unsigned short*)(ws + 1179648);           // 2*256*256 bf16 (p-major)
  unsigned short* Wob = (unsigned short*)(ws + 1179648 + 262144);  // 256*768 bf16
  float*          QKV = (float*)(ws + 1179648 + 262144 + 393216);  // 2*2304*256 f32
  unsigned short* AOT = (unsigned short*)(ws + 1179648 + 262144 + 393216 + 4718592); // 2*256*768 bf16

  convert_bf16<<<(kO3 * kC + 255) / 256, 256, 0, stream>>>(wqkv, Wb, kO3 * kC);
  convert_x_T<<<(kB * kC * kHW) / 256, 256, 0, stream>>>(x, XT);
  convert_bf16<<<(kC * kTD + 255) / 256, 256, 0, stream>>>(wout, Wob, kC * kTD);

  qkv_gemm<<<(kB * (kO3 / 16) * (kHW / 32)) / 8, 256, 0, stream>>>(Wb, XT, QKV);
  attn_softmax<<<kB * kTD, 256, 0, stream>>>(QKV, AOT);
  proj_norm<<<kB * 16, 256, 0, stream>>>(Wob, AOT, x, bout, y);
}
